// TrellisMoEMLP_84318797955744
// MI455X (gfx1250) — matrix operand order repaired and checked by reading, not compiled, not run on hardware
//
#include <hip/hip_runtime.h>

// ---------------- problem constants (from reference) ----------------
#define TT 4096
#define HH 1024
#define II 2048
#define EE 32
#define KK 4

typedef __attribute__((ext_vector_type(8)))  float  v8f;
typedef __attribute__((ext_vector_type(2)))  __bf16 bf16x2;
typedef __attribute__((ext_vector_type(4)))  __bf16 bf16x4;
typedef __attribute__((ext_vector_type(8)))  __bf16 bf16x8;
typedef __attribute__((ext_vector_type(16))) __bf16 bf16x16;

// FFN tiling: 32 tokens (2 m-tiles) per block, 8 waves
#define MT 32       // tokens per block
#define IC 512      // I-chunk held in LDS as bf16 hidden tile
#define KC 64       // K panel staged per wave per n-tile
#define BSTRIDE 72  // KC + 8 pad (keeps 16B alignment of fragment loads)
#define XSTRIDE (HH + 8)
#define HSTRIDE (IC + 8)

__device__ __forceinline__ v8f zero8() {
  v8f z = {0.f, 0.f, 0.f, 0.f, 0.f, 0.f, 0.f, 0.f};
  return z;
}

// A fragment (16x32 bf16): lane_lo = row M; lanes 0-15 hold K {kb..kb+7, kb+16..kb+23},
// lanes 16-31 hold K {kb+8..kb+15, kb+24..kb+31}  (ISA 7.12.2 16-bit A layout)
__device__ __forceinline__ bf16x16 load_fragA(const __bf16* row, int kb, int lane_hi) {
  bf16x8 lo = *(const bf16x8*)(row + kb + lane_hi * 8);
  bf16x8 hi = *(const bf16x8*)(row + kb + 16 + lane_hi * 8);
  return __builtin_shufflevector(lo, hi, 0,1,2,3,4,5,6,7,8,9,10,11,12,13,14,15);
}

// B fragment (32x16 bf16): lane_lo = col N; lanes 0-15 hold K {kb..kb+7, kb+16..kb+23},
// lanes 16-31 hold K {kb+8..kb+15, kb+24..kb+31} (same K order as A fragment)
__device__ __forceinline__ bf16x16 load_fragB(const __bf16* row, int kb, int lane_hi) {
  bf16x8 lo = *(const bf16x8*)(row + kb + lane_hi * 8);
  bf16x8 hi = *(const bf16x8*)(row + kb + 16 + lane_hi * 8);
  return __builtin_shufflevector(lo, hi, 0,1,2,3,4,5,6,7,8,9,10,11,12,13,14,15);
}

// Stage KC x 16 f32 weight panel -> bf16, transposed into wave-private LDS [n][k].
// Each lane loads two consecutive-K rows (coalesced b128 across lanes) and stores
// bf16 K-pairs (compiler fuses them into ds_store_2addr_b32).
// Prefetch: 2 global_prefetch_b8 per lane cover ALL 64 rows of the NEXT K-chunk,
// so its blocking loads hit L0/L2 one chunk later (no LOADcnt cost).
__device__ __forceinline__ void stage_b(__bf16* dst, const float* __restrict__ src,
                                        int src_stride, int k0, int n0, int lane) {
  const float* base = src + (size_t)k0 * src_stride + n0;
  __builtin_prefetch(base + (size_t)(KC + lane) * src_stride, 0, 1);
  __builtin_prefetch(base + (size_t)(KC + 32 + lane) * src_stride, 0, 1);
#pragma unroll
  for (int i = 0; i < 4; ++i) {
    int rp = i * 8 + (lane >> 2);  // row-pair 0..31
    int k = rp * 2;
    int n = (lane & 3) * 4;
    const float* p = base + (size_t)k * src_stride + n;
    float4 a = *(const float4*)p;
    float4 b = *(const float4*)(p + src_stride);
    bf16x2 p0 = {(__bf16)a.x, (__bf16)b.x};
    bf16x2 p1 = {(__bf16)a.y, (__bf16)b.y};
    bf16x2 p2 = {(__bf16)a.z, (__bf16)b.z};
    bf16x2 p3 = {(__bf16)a.w, (__bf16)b.w};
    *(bf16x2*)(dst + (n + 0) * BSTRIDE + k) = p0;
    *(bf16x2*)(dst + (n + 1) * BSTRIDE + k) = p1;
    *(bf16x2*)(dst + (n + 2) * BSTRIDE + k) = p2;
    *(bf16x2*)(dst + (n + 3) * BSTRIDE + k) = p3;
  }
}

// ---------------- kernel 1: cast x to bf16 ----------------
__global__ void cast_x_kernel(const float* __restrict__ x, __bf16* __restrict__ xb) {
  int i = blockIdx.x * blockDim.x + threadIdx.x;  // one float4 per thread
  float4 v = ((const float4*)x)[i];
  bf16x4 o = {(__bf16)v.x, (__bf16)v.y, (__bf16)v.z, (__bf16)v.w};
  ((bf16x4*)xb)[i] = o;
}

// ---------------- kernel 2: router (one wave32 per token) ----------------
__global__ void router_kernel(const float* __restrict__ x, const float* __restrict__ rw,
                              int* __restrict__ counts, int* __restrict__ toks,
                              float* __restrict__ wts) {
  int token = (blockIdx.x * blockDim.x + threadIdx.x) >> 5;
  int lane = threadIdx.x & 31;

  const float4* xr = (const float4*)(x + (size_t)token * HH);
  const float4* wr = (const float4*)(rw + (size_t)lane * HH);
  float acc = 0.f;
#pragma unroll 8
  for (int i = 0; i < HH / 4; ++i) {
    float4 a = xr[i], b = wr[i];
    acc += a.x * b.x + a.y * b.y + a.z * b.z + a.w * b.w;
  }
  // softmax over 32 experts (one per lane)
  float m = acc;
  for (int off = 16; off; off >>= 1) m = fmaxf(m, __shfl_xor(m, off));
  float p = __expf(acc - m);
  float s = p;
  for (int off = 16; off; off >>= 1) s += __shfl_xor(s, off);
  p /= s;

  // iterative top-4 with wave reductions
  float myp = p;
  int eidx[KK];
  float ew[KK];
#pragma unroll
  for (int j = 0; j < KK; ++j) {
    float v = myp;
    int idx = lane;
    for (int off = 16; off; off >>= 1) {
      float ov = __shfl_xor(v, off);
      int oi = __shfl_xor(idx, off);
      if (ov > v || (ov == v && oi < idx)) { v = ov; idx = oi; }
    }
    eidx[j] = idx;
    ew[j] = v;
    if (lane == idx) myp = -1.0f;
  }
  float s4 = ew[0] + ew[1] + ew[2] + ew[3];
  if (lane < KK) {
    int e = eidx[lane];
    int slot = atomicAdd(&counts[e], 1);
    toks[e * TT + slot] = token;
    wts[e * TT + slot] = ew[lane] / s4;
  }
}

// ---------------- kernel 3: fused SwiGLU FFN (gathered or dense) ----------------
// Block = 32 tokens (2 m-tiles) of one expert, 8 waves.
// Phase 1: h = silu(x@Wg)*(x@Wu) for an I-chunk -> LDS (bf16).
// Phase 2: out += h_chunk @ Wd_chunk accumulated in VGPRs across chunks.
// Each staged B panel is reused by both m-tiles (2x arithmetic intensity).
template <bool GATHER>
__global__ __launch_bounds__(256, 1) void ffn_kernel(
    const __bf16* __restrict__ xb, const float* __restrict__ gW,
    const float* __restrict__ uW, const float* __restrict__ dW,
    const int* __restrict__ counts, const int* __restrict__ toks,
    const float* __restrict__ wts, float* __restrict__ out) {
  __shared__ __bf16 xs[MT * XSTRIDE];          // 66 KB: gathered x tile (bf16)
  __shared__ __bf16 hs[MT * HSTRIDE];          // 33 KB: hidden chunk (bf16)
  __shared__ __bf16 Bbuf[8][2][16 * BSTRIDE];  // 37 KB: per-wave weight panels
  __shared__ int tok_s[MT];
  __shared__ float wt_s[MT];

  const float *gw, *uw, *dw;
  int e, tile, cnt;
  if (GATHER) {
    e = blockIdx.x >> 7;     // 128 max tiles per expert (T/MT)
    tile = blockIdx.x & 127;
    cnt = counts[e];
    if (tile * MT >= cnt) return;  // uniform per block
    gw = gW + (size_t)e * HH * II;
    uw = uW + (size_t)e * HH * II;
    dw = dW + (size_t)e * II * HH;
  } else {
    e = 0; tile = blockIdx.x; cnt = TT;
    gw = gW; uw = uW; dw = dW;
  }

  const int tid = threadIdx.x;
  const int lane = tid & 31;
  const int wave = tid >> 5;
  const int lane_lo = lane & 15;
  const int lane_hi = lane >> 4;

  if (tid < MT) {
    int slot = tile * MT + tid;
    int tk; float w;
    if (GATHER) {
      if (slot < cnt) { tk = toks[e * TT + slot]; w = wts[e * TT + slot]; }
      else            { tk = 0; w = 0.f; }
    } else { tk = slot; w = 1.f; }
    tok_s[tid] = tk;
    wt_s[tid] = w;
  }
  __syncthreads();

  // async-gather 32 token rows of x (bf16) into LDS: 8 threads/row, 128 elems each,
  // issued as ASYNCcnt-tracked direct global->LDS DMA loads.
  {
    int r = tid >> 3;
    int c0 = (tid & 7) * 128;
    const __bf16* src = xb + (size_t)tok_s[r] * HH + c0;
    unsigned dst = (unsigned)(uintptr_t)(xs + r * XSTRIDE + c0);
#pragma unroll
    for (int i = 0; i < 16; ++i) {
      asm volatile("global_load_async_to_lds_b128 %0, %1, off"
                   :: "v"(dst + i * 16), "v"(src + i * 8)
                   : "memory");
    }
    asm volatile("s_wait_asynccnt 0x0" ::: "memory");
  }
  __syncthreads();

  v8f acc_o[16];  // [n-tile 0..7][m-tile 0..1]
#pragma unroll
  for (int t = 0; t < 16; ++t) acc_o[t] = zero8();

  for (int c = 0; c < II / IC; ++c) {  // 4 chunks over I
    // -------- phase 1: hidden chunk --------
#pragma unroll 1
    for (int t = 0; t < (IC / 16) / 8; ++t) {  // 4 n-tiles per wave
      int nt_local = wave * 4 + t;             // 0..31 within chunk
      int n0 = c * IC + nt_local * 16;         // global I column
      v8f ag[2] = {zero8(), zero8()};
      v8f au[2] = {zero8(), zero8()};
      for (int k0 = 0; k0 < HH; k0 += KC) {
        stage_b(Bbuf[wave][0], gw, II, k0, n0, lane);
        stage_b(Bbuf[wave][1], uw, II, k0, n0, lane);
#pragma unroll
        for (int kk = 0; kk < KC; kk += 32) {
          bf16x16 bg = load_fragB(Bbuf[wave][0] + lane_lo * BSTRIDE, kk, lane_hi);
          bf16x16 bu = load_fragB(Bbuf[wave][1] + lane_lo * BSTRIDE, kk, lane_hi);
#pragma unroll
          for (int m = 0; m < 2; ++m) {
            bf16x16 a = load_fragA(xs + (m * 16 + lane_lo) * XSTRIDE, k0 + kk, lane_hi);
            ag[m] = __builtin_amdgcn_wmma_f32_16x16x32_bf16(false, a, false, bg,
                                                            (short)0, ag[m], false, false);
            au[m] = __builtin_amdgcn_wmma_f32_16x16x32_bf16(false, a, false, bu,
                                                            (short)0, au[m], false, false);
          }
        }
      }
      // SwiGLU (fast rcp sigmoid) -> bf16 hidden tile
      int hcol = nt_local * 16 + lane_lo;
#pragma unroll
      for (int m = 0; m < 2; ++m) {
#pragma unroll
        for (int v = 0; v < 8; ++v) {
          float g = ag[m][v], u = au[m][v];
          float hval = g * __builtin_amdgcn_rcpf(1.f + __expf(-g)) * u;
          hs[(m * 16 + v + 8 * lane_hi) * HSTRIDE + hcol] = (__bf16)hval;
        }
      }
    }
    __syncthreads();

    // -------- phase 2: out += h_chunk @ Wd_chunk --------
    const float* dwc = dw + (size_t)(c * IC) * HH;
#pragma unroll 1
    for (int t = 0; t < 8; ++t) {  // 8 H n-tiles per wave (8*8*16 = 1024)
      int n0 = (wave * 8 + t) * 16;
      v8f acc0 = acc_o[t * 2 + 0];
      v8f acc1 = acc_o[t * 2 + 1];
      for (int k0 = 0; k0 < IC; k0 += KC) {
        stage_b(Bbuf[wave][0], dwc, HH, k0, n0, lane);
#pragma unroll
        for (int kk = 0; kk < KC; kk += 32) {
          bf16x16 b = load_fragB(Bbuf[wave][0] + lane_lo * BSTRIDE, kk, lane_hi);
          bf16x16 a0 = load_fragA(hs + lane_lo * HSTRIDE, k0 + kk, lane_hi);
          bf16x16 a1 = load_fragA(hs + (16 + lane_lo) * HSTRIDE, k0 + kk, lane_hi);
          acc0 = __builtin_amdgcn_wmma_f32_16x16x32_bf16(false, a0, false, b,
                                                         (short)0, acc0, false, false);
          acc1 = __builtin_amdgcn_wmma_f32_16x16x32_bf16(false, a1, false, b,
                                                         (short)0, acc1, false, false);
        }
      }
      acc_o[t * 2 + 0] = acc0;
      acc_o[t * 2 + 1] = acc1;
    }
    __syncthreads();
  }

  // -------- epilogue: scatter --------
#pragma unroll 1
  for (int t = 0; t < 8; ++t) {
    int col = (wave * 8 + t) * 16 + lane_lo;
#pragma unroll
    for (int m = 0; m < 2; ++m) {
#pragma unroll
      for (int v = 0; v < 8; ++v) {
        int row = m * 16 + v + 8 * lane_hi;
        float w = wt_s[row];
        float val = acc_o[t * 2 + m][v];
        if (GATHER) {
          if (w != 0.f) atomicAdd(out + (size_t)tok_s[row] * HH + col, val * w);
        } else {
          out[(size_t)tok_s[row] * HH + col] = val;  // shared expert writes first
        }
      }
    }
  }
}

// ---------------- host launch ----------------
extern "C" void kernel_launch(void* const* d_in, const int* in_sizes, int n_in,
                              void* d_out, int out_size, void* d_ws, size_t ws_size,
                              hipStream_t stream) {
  const float* x  = (const float*)d_in[0];
  const float* rw = (const float*)d_in[1];
  const float* gW = (const float*)d_in[2];
  const float* uW = (const float*)d_in[3];
  const float* dW = (const float*)d_in[4];
  const float* sg = (const float*)d_in[5];
  const float* su = (const float*)d_in[6];
  const float* sd = (const float*)d_in[7];
  float* out = (float*)d_out;

  // workspace layout
  char* ws = (char*)d_ws;
  __bf16* xb  = (__bf16*)ws;        // T*H*2 = 8 MB
  size_t off  = (size_t)TT * HH * 2;
  int* counts = (int*)(ws + off);   // 128 B (+pad)
  off += 1024;
  int* toks   = (int*)(ws + off);   // E*T*4
  off += (size_t)EE * TT * 4;
  float* wts  = (float*)(ws + off); // E*T*4

  hipMemsetAsync(counts, 0, EE * sizeof(int), stream);

  cast_x_kernel<<<(TT * HH / 4) / 256, 256, 0, stream>>>(x, xb);

  router_kernel<<<TT / 8, 256, 0, stream>>>(x, rw, counts, toks, wts);

  // shared expert: dense over all tokens, plain store initializes d_out
  ffn_kernel<false><<<TT / MT, 256, 0, stream>>>(xb, sg, su, sd,
                                                 counts, toks, wts, out);

  // routed experts: gather + weighted atomic accumulate
  ffn_kernel<true><<<EE * 128, 256, 0, stream>>>(xb, gW, uW, dW,
                                                 counts, toks, wts, out);
}
